// SDSG32_3496103379551
// MI455X (gfx1250) — matrix operand order repaired and checked by reading, not compiled, not run on hardware
//
#include <hip/hip_runtime.h>
#include <hip/hip_bf16.h>

typedef __attribute__((ext_vector_type(16))) _Float16 v16h;
typedef __attribute__((ext_vector_type(8)))  _Float16 v8h;
typedef __attribute__((ext_vector_type(8)))  float    v8f;

#define IN_C 128
#define HID  32
#define OUTC 64

// ---------------------------------------------------------------- helpers

__device__ __forceinline__ v8f wmma16(v16h a, v16h b, v8f c) {
  // D = A(16x32 f16) * B(32x16 f16) + C(16x16 f32)
  return __builtin_amdgcn_wmma_f32_16x16x32_f16(
      /*neg_a=*/false, a, /*neg_b=*/false, b,
      /*c_mod=*/(short)0, c, /*reuse_a=*/false, /*reuse_b=*/false);
}

// Load 16 contiguous f16 as one B fragment: lane holds K = 0..15 (+16 for hi half-wave
// handled by caller's pointer offset), col handled by caller's row offset.
__device__ __forceinline__ v16h ldb16(const _Float16* __restrict__ p) {
  v8h lo = *(const v8h*)p;
  v8h hi = *(const v8h*)(p + 8);
  v16h r;
#pragma unroll
  for (int j = 0; j < 8; ++j) { r[j] = lo[j]; r[j + 8] = hi[j]; }
  return r;
}

// A fragment from a row-major f32 row: cols [abase,abase+16)
__device__ __forceinline__ v16h lda_f32(const float* __restrict__ rowptr, int abase) {
  const float4* q0 = (const float4*)(rowptr + abase);
  const float4* q1 = (const float4*)(rowptr + abase + 8);
  float4 u0 = q0[0], u1 = q0[1], u2 = q1[0], u3 = q1[1];
  v16h a;
  a[0]=(_Float16)u0.x; a[1]=(_Float16)u0.y; a[2]=(_Float16)u0.z; a[3]=(_Float16)u0.w;
  a[4]=(_Float16)u1.x; a[5]=(_Float16)u1.y; a[6]=(_Float16)u1.z; a[7]=(_Float16)u1.w;
  a[8]=(_Float16)u2.x; a[9]=(_Float16)u2.y; a[10]=(_Float16)u2.z; a[11]=(_Float16)u2.w;
  a[12]=(_Float16)u3.x; a[13]=(_Float16)u3.y; a[14]=(_Float16)u3.z; a[15]=(_Float16)u3.w;
  return a;
}

// min/max reduce across the 16 lanes of a half-wave (xor masks 1,2,4,8 stay in-half)
__device__ __forceinline__ float rmin16(float v) {
#pragma unroll
  for (int m = 1; m < 16; m <<= 1) v = fminf(v, __shfl_xor(v, m, 32));
  return v;
}
__device__ __forceinline__ float rmax16(float v) {
#pragma unroll
  for (int m = 1; m < 16; m <<= 1) v = fmaxf(v, __shfl_xor(v, m, 32));
  return v;
}

// ---------------------------------------------------------------- small kernels

__global__ void k_cvt_f16(const float* __restrict__ s, _Float16* __restrict__ d, int n) {
  int i = blockIdx.x * blockDim.x + threadIdx.x;
  if (i < n) d[i] = (_Float16)s[i];
}

__global__ void k_deg_init(float* __restrict__ deg, int n) {
  int i = blockIdx.x * blockDim.x + threadIdx.x;
  if (i < n) deg[i] = 1.0f;  // self loop
}

__global__ void k_deg_count(const int* __restrict__ col, float* __restrict__ deg, int e) {
  int i = blockIdx.x * blockDim.x + threadIdx.x;
  if (i < e) atomicAdd(&deg[col[i]], 1.0f);
}

__global__ void k_rsqrt(float* __restrict__ deg, int n) {
  int i = blockIdx.x * blockDim.x + threadIdx.x;
  if (i < n) deg[i] = __frsqrt_rn(deg[i]);  // deg >= 1 always
}

__global__ void k_nrm(const int* __restrict__ row, const int* __restrict__ col,
                      const float* __restrict__ dis, float* __restrict__ nrm, int e) {
  int i = blockIdx.x * blockDim.x + threadIdx.x;
  if (i < e) nrm[i] = dis[row[i]] * dis[col[i]];
}

// p = dis^2 * h   (self-loop contribution initializes the scatter target)
__global__ void k_prop_init(const float* __restrict__ dis, const float* __restrict__ h,
                            float* __restrict__ p, int n32) {
  int i = blockIdx.x * blockDim.x + threadIdx.x;
  if (i < n32) {
    float d = dis[i >> 5];
    p[i] = d * d * h[i];
  }
}

// Edge propagation: each wave owns batches of 32 edges. Each lane loads its own
// (row,col,nrm) triple coalesced, then the triples are broadcast one edge at a
// time with __shfl (lane-permute); all 32 lanes do the coalesced 128B feature
// gather h[row][*] and 128B atomic-add burst into p[col][*]. Edge streams and
// the 12.8MB feature matrices stay L2-resident across layers.
__global__ void k_prop_edges(const int* __restrict__ row, const int* __restrict__ col,
                             const float* __restrict__ nrm, const float* __restrict__ h,
                             float* __restrict__ p, int e) {
  int lane = threadIdx.x & 31;
  int wid = (blockIdx.x * blockDim.x + threadIdx.x) >> 5;
  int nw = (gridDim.x * blockDim.x) >> 5;
  for (int base = wid << 5; base < e; base += (nw << 5)) {
    int idx = base + lane;
    bool ok = idx < e;
    int r = ok ? row[idx] : 0;
    int c = ok ? col[idx] : -1;
    float w = ok ? nrm[idx] : 0.0f;
#pragma unroll 4
    for (int j = 0; j < 32; ++j) {
      int cj = __shfl(c, j, 32);
      if (cj < 0) break;                 // wave-uniform tail condition
      int rj = __shfl(r, j, 32);
      float wj = __shfl(w, j, 32);
      atomicAdd(&p[(size_t)cj * HID + lane], wj * h[(size_t)rj * HID + lane]);
    }
  }
}

// ---------------------------------------------------------------- WMMA kernels

// x0 = mynorm(relu(x @ W1^T + b1)); store x0 (f32), x0 raw (f16), m[0]=mynorm(x0) (f16)
__global__ void k_x0(const float* __restrict__ x, const _Float16* __restrict__ w1h,
                     const float* __restrict__ b1, float* __restrict__ h,
                     _Float16* __restrict__ m0dst, _Float16* __restrict__ rawdst, int N) {
  int wave = (blockIdx.x * blockDim.x + threadIdx.x) >> 5;
  int lane = threadIdx.x & 31;
  int row0 = wave << 4;
  if (row0 >= N) return;
  int half = lane >> 4, ln = lane & 15;
  int abase = half * 16;
  int arow = row0 + ln; if (arow >= N) arow = N - 1;
  const float* xr = x + (size_t)arow * IN_C;

  v8f c0 = {}, c1 = {};
#pragma unroll
  for (int kc = 0; kc < 4; ++kc) {
    v16h a = lda_f32(xr + kc * 32, abase);
    v16h b0 = ldb16(w1h + (size_t)ln * IN_C + kc * 32 + half * 16);
    v16h b1t = ldb16(w1h + (size_t)(ln + 16) * IN_C + kc * 32 + half * 16);
    c0 = wmma16(a, b0, c0);
    c1 = wmma16(a, b1t, c1);
  }
  float bb0 = b1[ln], bb1 = b1[ln + 16];
  float t0[8], t1[8];
#pragma unroll
  for (int r = 0; r < 8; ++r) {
    t0[r] = fmaxf(c0[r] + bb0, 0.0f);
    t1[r] = fmaxf(c1[r] + bb1, 0.0f);
  }
  // mynorm #1 -> x0
#pragma unroll
  for (int r = 0; r < 8; ++r) {
    float mn = rmin16(fminf(t0[r], t1[r]));
    float mx = rmax16(fmaxf(t0[r], t1[r]));
    float s = 2.0f / (mx - mn + 1e-8f);
    t0[r] = s * (t0[r] - mn) - 1.0f;
    t1[r] = s * (t1[r] - mn) - 1.0f;
  }
#pragma unroll
  for (int r = 0; r < 8; ++r) {
    int rr = row0 + r + half * 8;
    if (rr < N) {
      size_t o = (size_t)rr * HID + ln;
      h[o] = t0[r];               h[o + 16] = t1[r];
      rawdst[o] = (_Float16)t0[r]; rawdst[o + 16] = (_Float16)t1[r];
    }
  }
  // mynorm #2 -> m[0]
#pragma unroll
  for (int r = 0; r < 8; ++r) {
    float mn = rmin16(fminf(t0[r], t1[r]));
    float mx = rmax16(fmaxf(t0[r], t1[r]));
    float s = 2.0f / (mx - mn + 1e-8f);
    int rr = row0 + r + half * 8;
    if (rr < N) {
      size_t o = (size_t)rr * HID + ln;
      m0dst[o]      = (_Float16)(s * (t0[r] - mn) - 1.0f);
      m0dst[o + 16] = (_Float16)(s * (t1[r] - mn) - 1.0f);
    }
  }
}

// y = p @ Wc[l]^T + bc[l]; store y (f32 next-h), m[l+1] (f16), optional raw f16
__global__ void k_layer(const float* __restrict__ pin, const _Float16* __restrict__ w,
                        const float* __restrict__ bias, float* __restrict__ hn,
                        _Float16* __restrict__ mdst, _Float16* __restrict__ rawdst, int N) {
  int wave = (blockIdx.x * blockDim.x + threadIdx.x) >> 5;
  int lane = threadIdx.x & 31;
  int row0 = wave << 4;
  if (row0 >= N) return;
  int half = lane >> 4, ln = lane & 15;
  int abase = half * 16;
  int arow = row0 + ln; if (arow >= N) arow = N - 1;

  v16h a = lda_f32(pin + (size_t)arow * HID, abase);
  v16h b0 = ldb16(w + (size_t)ln * HID + half * 16);
  v16h b1t = ldb16(w + (size_t)(ln + 16) * HID + half * 16);
  v8f c0 = {}, c1 = {};
  c0 = wmma16(a, b0, c0);
  c1 = wmma16(a, b1t, c1);

  float bb0 = bias[ln], bb1 = bias[ln + 16];
  float t0[8], t1[8];
#pragma unroll
  for (int r = 0; r < 8; ++r) { t0[r] = c0[r] + bb0; t1[r] = c1[r] + bb1; }

#pragma unroll
  for (int r = 0; r < 8; ++r) {
    int rr = row0 + r + half * 8;
    if (rr < N) {
      size_t o = (size_t)rr * HID + ln;
      hn[o] = t0[r]; hn[o + 16] = t1[r];
      if (rawdst) { rawdst[o] = (_Float16)t0[r]; rawdst[o + 16] = (_Float16)t1[r]; }
    }
  }
#pragma unroll
  for (int r = 0; r < 8; ++r) {
    float mn = rmin16(fminf(t0[r], t1[r]));
    float mx = rmax16(fmaxf(t0[r], t1[r]));
    float s = 2.0f / (mx - mn + 1e-8f);
    int rr = row0 + r + half * 8;
    if (rr < N) {
      size_t o = (size_t)rr * HID + ln;
      mdst[o]      = (_Float16)(s * (t0[r] - mn) - 1.0f);
      mdst[o + 16] = (_Float16)(s * (t1[r] - mn) - 1.0f);
    }
  }
}

// out = cat @ W32^T + b32, with cat's 32-col chunk k built on the fly:
//   k==0 -> raw x0 (slot 32), k==1 -> raw xs1 (slot 33), k==16 -> raw xs16 (slot 34),
//   else m[k]-m[k-2] (slots k, k-2), all f16.
__global__ void k_final(const _Float16* __restrict__ mbuf, const _Float16* __restrict__ w32h,
                        const float* __restrict__ b32, float* __restrict__ out, int N) {
  int wave = (blockIdx.x * blockDim.x + threadIdx.x) >> 5;
  int lane = threadIdx.x & 31;
  int row0 = wave << 4;
  if (row0 >= N) return;
  int half = lane >> 4, ln = lane & 15;
  int abase = half * 16;
  int arow = row0 + ln; if (arow >= N) arow = N - 1;

  size_t NS = (size_t)N * HID;
  size_t roff = (size_t)arow * HID + abase;

  v8f acc[4] = {};
  for (int k = 0; k < 32; ++k) {
    v8h lo, hi;
    if (k == 0 || k == 1 || k == 16) {
      int slot = (k == 0) ? 32 : (k == 1) ? 33 : 34;
      const _Float16* s = mbuf + (size_t)slot * NS + roff;
      lo = *(const v8h*)s;
      hi = *(const v8h*)(s + 8);
    } else {
      const _Float16* s1 = mbuf + (size_t)k * NS + roff;
      const _Float16* s0 = mbuf + (size_t)(k - 2) * NS + roff;
      lo = *(const v8h*)s1 - *(const v8h*)s0;
      hi = *(const v8h*)(s1 + 8) - *(const v8h*)(s0 + 8);
    }
    v16h a;
#pragma unroll
    for (int j = 0; j < 8; ++j) { a[j] = lo[j]; a[j + 8] = hi[j]; }
#pragma unroll
    for (int t = 0; t < 4; ++t) {
      v16h b = ldb16(w32h + (size_t)(ln + 16 * t) * 1024 + k * 32 + half * 16);
      acc[t] = wmma16(a, b, acc[t]);
    }
  }
#pragma unroll
  for (int t = 0; t < 4; ++t) {
    float bb = b32[ln + 16 * t];
#pragma unroll
    for (int r = 0; r < 8; ++r) {
      int rr = row0 + r + half * 8;
      if (rr < N) out[(size_t)rr * OUTC + ln + 16 * t] = acc[t][r] + bb;
    }
  }
}

// ---------------------------------------------------------------- launch

extern "C" void kernel_launch(void* const* d_in, const int* in_sizes, int n_in,
                              void* d_out, int out_size, void* d_ws, size_t ws_size,
                              hipStream_t stream) {
  const float* x   = (const float*)d_in[0];
  const int*   ei  = (const int*)d_in[1];
  const float* W1  = (const float*)d_in[2];
  const float* b1  = (const float*)d_in[3];
  const float* Wc  = (const float*)d_in[4];
  const float* bc  = (const float*)d_in[5];
  const float* W32 = (const float*)d_in[6];
  const float* b32 = (const float*)d_in[7];

  const int N = in_sizes[0] / IN_C;
  const int E = in_sizes[1] / 2;
  const int L = in_sizes[4] / (HID * HID);
  const int* erow = ei;
  const int* ecol = ei + E;
  const size_t NS = (size_t)N * HID;

  char* base = (char*)d_ws;
  auto alloc = [&](size_t bytes) {
    char* r = base;
    base += (bytes + 255) & ~(size_t)255;
    return r;
  };
  float*    dis  = (float*)alloc((size_t)N * 4);        // deg, then rsqrt(deg)
  float*    nrm  = (float*)alloc((size_t)E * 4);
  float*    h0   = (float*)alloc(NS * 4);
  float*    h1   = (float*)alloc(NS * 4);
  float*    pb   = (float*)alloc(NS * 4);
  _Float16* mbuf = (_Float16*)alloc(NS * 2 * 35);       // 0..31: m[k]; 32..34: raw x0/x1/x16
  _Float16* w1h  = (_Float16*)alloc((size_t)HID * IN_C * 2);
  _Float16* wch  = (_Float16*)alloc((size_t)L * HID * HID * 2);
  _Float16* w32h = (_Float16*)alloc((size_t)OUTC * 1024 * 2);
  (void)ws_size; (void)n_in; (void)out_size;

  k_cvt_f16<<<(HID * IN_C + 255) / 256, 256, 0, stream>>>(W1, w1h, HID * IN_C);
  k_cvt_f16<<<(L * HID * HID + 255) / 256, 256, 0, stream>>>(Wc, wch, L * HID * HID);
  k_cvt_f16<<<(OUTC * 1024 + 255) / 256, 256, 0, stream>>>(W32, w32h, OUTC * 1024);

  k_deg_init<<<(N + 255) / 256, 256, 0, stream>>>(dis, N);
  k_deg_count<<<(E + 255) / 256, 256, 0, stream>>>(ecol, dis, E);
  k_rsqrt<<<(N + 255) / 256, 256, 0, stream>>>(dis, N);
  k_nrm<<<(E + 255) / 256, 256, 0, stream>>>(erow, ecol, dis, nrm, E);

  const int waves = (N + 15) / 16;
  const int gblocks = (waves + 7) / 8;   // 8 waves (256 threads) per block

  k_x0<<<gblocks, 256, 0, stream>>>(x, w1h, b1, h0, mbuf /*m[0]*/, mbuf + 32 * NS, N);

  float* hc = h0;
  float* hn = h1;
  for (int l = 0; l < L; ++l) {
    k_prop_init<<<((int)NS + 255) / 256, 256, 0, stream>>>(dis, hc, pb, (int)NS);
    k_prop_edges<<<2048, 256, 0, stream>>>(erow, ecol, nrm, hc, pb, E);
    _Float16* raw = (l == 0) ? (mbuf + 33 * NS)
                  : (l == 15) ? (mbuf + 34 * NS)
                  : (_Float16*)nullptr;
    k_layer<<<gblocks, 256, 0, stream>>>(pb, wch + (size_t)l * HID * HID,
                                         bc + (size_t)l * HID, hn,
                                         mbuf + (size_t)(l + 1) * NS, raw, N);
    float* t = hc; hc = hn; hn = t;
  }

  k_final<<<gblocks, 256, 0, stream>>>(mbuf, w32h, b32, (float*)d_out, N);
}
